// ScaledDotAttention_16716012716549
// MI455X (gfx1250) — matrix operand order repaired and checked by reading, not compiled, not run on hardware
//
#include <hip/hip_runtime.h>
#include <hip/hip_bf16.h>

typedef __attribute__((ext_vector_type(16))) __bf16 v16bf;
typedef __attribute__((ext_vector_type(8)))  __bf16 v8bf;
typedef __attribute__((ext_vector_type(8)))  float  v8f;

static constexpr int BATCH = 4;
static constexpr int SEQ   = 2048;
static constexpr int HID   = 1024;

static constexpr int BM = 128, BN = 128, BK = 32;
static constexpr int LDS_STRIDE = 40; // 32 bf16 + 8 pad -> 80B rows, 16B aligned, bank-conflict free

__device__ __forceinline__ unsigned short f2bf(float f) {
  unsigned int u = __float_as_uint(f);
  u += 0x7fffu + ((u >> 16) & 1u);          // round-to-nearest-even
  return (unsigned short)(u >> 16);
}

// Synchronous stage of a 128x32 tile into LDS with f32 -> bf16 conversion.
__device__ __forceinline__ void stage_tile_f32(const float* __restrict__ g, int ld,
                                               unsigned short* __restrict__ lds, int tid) {
#pragma unroll
  for (int it = 0; it < 2; ++it) {
    const int c   = tid + it * 256;   // chunk id 0..511, 8 elements each
    const int row = c >> 2;
    const int kc  = (c & 3) * 8;
    unsigned short* dst = lds + row * LDS_STRIDE + kc;
    const float* src = g + (size_t)row * ld + kc;
    const float4 f0 = *(const float4*)(src);
    const float4 f1 = *(const float4*)(src + 4);
    uint4 p;
    p.x = (unsigned)f2bf(f0.x) | ((unsigned)f2bf(f0.y) << 16);
    p.y = (unsigned)f2bf(f0.z) | ((unsigned)f2bf(f0.w) << 16);
    p.z = (unsigned)f2bf(f1.x) | ((unsigned)f2bf(f1.y) << 16);
    p.w = (unsigned)f2bf(f1.z) | ((unsigned)f2bf(f1.w) << 16);
    *(uint4*)dst = p;
  }
}

// Async stage (bf16 source): GLOBAL_LOAD_ASYNC_TO_LDS_B128 — memory->LDS without
// touching VGPRs, tracked on ASYNCcnt. LDS address = low 32 bits of generic pointer
__device__ __forceinline__ void stage_tile_async(const unsigned short* __restrict__ g, int ld,
                                                 unsigned short* __restrict__ lds, int tid) {
#pragma unroll
  for (int it = 0; it < 2; ++it) {
    const int c   = tid + it * 256;
    const int row = c >> 2;
    const int kc  = (c & 3) * 8;
    const unsigned int laddr =
        (unsigned int)(unsigned long long)(lds + row * LDS_STRIDE + kc);
    const unsigned short* gaddr = g + (size_t)row * ld + kc;
    asm volatile("global_load_async_to_lds_b128 %0, %1, off"
                 :
                 : "v"(laddr), "v"(gaddr)
                 : "memory");
  }
}

__device__ __forceinline__ void wait_async_lds() {
  asm volatile("s_wait_asynccnt 0x0" ::: "memory");
}

// A fragment (16x32 bf16, ISA layout): lane m holds row m; half 0 -> K 0-7 & 16-23,
// half 1 -> K 8-15 & 24-31.
__device__ __forceinline__ v16bf frag_a(const unsigned short* tile, int lane) {
  const int row = lane & 15, half = lane >> 4;
  const unsigned short* p = tile + row * LDS_STRIDE;
  v8bf lo = *(const v8bf*)(p + half * 8);
  v8bf hi = *(const v8bf*)(p + 16 + half * 8);
  return __builtin_shufflevector(lo, hi, 0,1,2,3,4,5,6,7,8,9,10,11,12,13,14,15);
}

// B fragment (32x16 bf16): lane n holds column n; half 0 -> K 0-7 & 16-23, half 1 -> K 8-15 & 24-31.
__device__ __forceinline__ v16bf frag_b(const unsigned short* tile, int lane) {
  const int col = lane & 15, half = lane >> 4;
  const unsigned short* p = tile + col * LDS_STRIDE + half * 8;
  v8bf lo = *(const v8bf*)(p);
  v8bf hi = *(const v8bf*)(p + 16);
  return __builtin_shufflevector(lo, hi, 0,1,2,3,4,5,6,7,8,9,10,11,12,13,14,15);
}

// Generic C[M,N] = A[M,K] * Bt[N,K]^T (+bias) with per-z-slice batching.
// MODE 0: f32 output, scaled.  MODE 1: bf16 output + bias.  MODE 2: bf16 transposed ([b][n][l]) + bias.
template<typename TA, typename TB, int MODE>
__global__ __launch_bounds__(256) void gemm_abT_kernel(
    const TA* __restrict__ A, const TB* __restrict__ Bt,
    const float* __restrict__ bias, float scale, void* __restrict__ outv,
    int K, int lda, int ldb, int ldo,
    long sAz, long sBz, long sOz)
{
  __shared__ __align__(16) unsigned short As[2][BM * LDS_STRIDE];
  __shared__ __align__(16) unsigned short Bs[2][BN * LDS_STRIDE];
  constexpr bool ASYNC_A = (sizeof(TA) == 2);
  constexpr bool ASYNC_B = (sizeof(TB) == 2);

  const int tid  = threadIdx.x;
  const int lane = tid & 31, wave = tid >> 5;
  const int wm = wave >> 2, wn = wave & 3;        // 2x4 wave grid, wave tile 64M x 32N
  const int mBlk = blockIdx.y * BM, nBlk = blockIdx.x * BN;
  const int z = blockIdx.z;

  const TA* Ag = A  + (size_t)z * sAz + (size_t)mBlk * lda;
  const TB* Bg = Bt + (size_t)z * sBz + (size_t)nBlk * ldb;

  v8f acc[4][2] = {};

  // Prologue: stage k-step 0 into buffer 0.
  if constexpr (ASYNC_A) stage_tile_async((const unsigned short*)Ag, lda, As[0], tid);
  else                   stage_tile_f32((const float*)Ag, lda, As[0], tid);
  if constexpr (ASYNC_B) stage_tile_async((const unsigned short*)Bg, ldb, Bs[0], tid);
  else                   stage_tile_f32((const float*)Bg, ldb, Bs[0], tid);
  if constexpr (ASYNC_A || ASYNC_B) wait_async_lds();
  __syncthreads();

  const int nsteps = K / BK;
  for (int s = 0; s < nsteps; ++s) {
    const int cur  = s & 1;
    const bool more = (s + 1) < nsteps;

    // Stage next k-step into the alternate buffer while computing on the current one.
    if (more) {
      const int k0 = (s + 1) * BK;
      if constexpr (ASYNC_A) stage_tile_async((const unsigned short*)Ag + k0, lda, As[cur ^ 1], tid);
      else                   stage_tile_f32((const float*)Ag + k0, lda, As[cur ^ 1], tid);
      if constexpr (ASYNC_B) stage_tile_async((const unsigned short*)Bg + k0, ldb, Bs[cur ^ 1], tid);
      else                   stage_tile_f32((const float*)Bg + k0, ldb, Bs[cur ^ 1], tid);
    }

    v16bf af[4], bfr[2];
#pragma unroll
    for (int mi = 0; mi < 4; ++mi)
      af[mi] = frag_a(As[cur] + (wm * 64 + mi * 16) * LDS_STRIDE, lane);
#pragma unroll
    for (int ni = 0; ni < 2; ++ni)
      bfr[ni] = frag_b(Bs[cur] + (wn * 32 + ni * 16) * LDS_STRIDE, lane);

#pragma unroll
    for (int mi = 0; mi < 4; ++mi)
#pragma unroll
      for (int ni = 0; ni < 2; ++ni)
        acc[mi][ni] = __builtin_amdgcn_wmma_f32_16x16x32_bf16(
            false, af[mi], false, bfr[ni], (short)0, acc[mi][ni], false, false);

    if (more) {
      if constexpr (ASYNC_A || ASYNC_B) wait_async_lds();
      __syncthreads();
    }
  }

  // C/D layout: lane -> n = lane&15, VGPR r -> m = r + 8*(lane>>4)
  const int half = lane >> 4;
  const int nl   = lane & 15;
#pragma unroll
  for (int mi = 0; mi < 4; ++mi) {
#pragma unroll
    for (int ni = 0; ni < 2; ++ni) {
      const int n  = nBlk + wn * 32 + ni * 16 + nl;
      const float bn = (MODE != 0) ? bias[n] : 0.0f;
#pragma unroll
      for (int r = 0; r < 8; ++r) {
        const int m = mBlk + wm * 64 + mi * 16 + r + 8 * half;
        const float v = acc[mi][ni][r];
        if constexpr (MODE == 0) {
          float* out = (float*)outv + (size_t)z * sOz;
          out[(size_t)m * ldo + n] = v * scale;
        } else if constexpr (MODE == 1) {
          unsigned short* out = (unsigned short*)outv;
          out[(size_t)m * ldo + n] = f2bf(v + bn);
        } else { // MODE 2: transposed V projection -> Vt[b][n][l]
          unsigned short* out = (unsigned short*)outv;
          const int b = m >> 11;            // m / SEQ
          const int l = m & (SEQ - 1);
          out[(size_t)b * HID * SEQ + (size_t)n * SEQ + l] = f2bf(v + bn);
        }
      }
    }
  }
}

// Row softmax over SEQ=2048 f32 scores (in place) + bf16 copy of probabilities.
__global__ __launch_bounds__(256) void softmax_kernel(float* __restrict__ scores,
                                                      unsigned short* __restrict__ pbf)
{
  __shared__ float red[8];
  const int row = blockIdx.x;
  float* p = scores + (size_t)row * SEQ;
  unsigned short* q = pbf + (size_t)row * SEQ;
  const int t = threadIdx.x;
  const int lane = t & 31, wave = t >> 5;

  float v[8];
  float mx = -3.402823466e38f;
#pragma unroll
  for (int i = 0; i < 8; ++i) { v[i] = p[t + i * 256]; mx = fmaxf(mx, v[i]); }
#pragma unroll
  for (int off = 16; off > 0; off >>= 1) mx = fmaxf(mx, __shfl_xor(mx, off, 32));
  if (lane == 0) red[wave] = mx;
  __syncthreads();
  float bm = red[0];
#pragma unroll
  for (int w = 1; w < 8; ++w) bm = fmaxf(bm, red[w]);

  float s = 0.0f;
#pragma unroll
  for (int i = 0; i < 8; ++i) { v[i] = __expf(v[i] - bm); s += v[i]; }
#pragma unroll
  for (int off = 16; off > 0; off >>= 1) s += __shfl_xor(s, off, 32);
  __syncthreads();
  if (lane == 0) red[wave] = s;
  __syncthreads();
  float tot = 0.0f;
#pragma unroll
  for (int w = 0; w < 8; ++w) tot += red[w];
  const float inv = 1.0f / tot;
#pragma unroll
  for (int i = 0; i < 8; ++i) {
    const float pv = v[i] * inv;
    p[t + i * 256] = pv;
    q[t + i * 256] = f2bf(pv);
  }
}

extern "C" void kernel_launch(void* const* d_in, const int* in_sizes, int n_in,
                              void* d_out, int out_size, void* d_ws, size_t ws_size,
                              hipStream_t stream) {
  (void)in_sizes; (void)n_in; (void)out_size; (void)ws_size;
  const float* queries = (const float*)d_in[0];
  const float* keys    = (const float*)d_in[1];
  const float* values  = (const float*)d_in[2];
  const float* Wq = (const float*)d_in[3];
  const float* bq = (const float*)d_in[4];
  const float* Wk = (const float*)d_in[5];
  const float* bk = (const float*)d_in[6];
  const float* Wv = (const float*)d_in[7];
  const float* bv = (const float*)d_in[8];

  constexpr size_t BLH = (size_t)BATCH * SEQ * HID;   // 8.39M
  constexpr size_t BLL = (size_t)BATCH * SEQ * SEQ;   // 16.8M

  float* ctx   = (float*)d_out;        // [B, L, H] f32
  float* attnF = ctx + BLH;            // [B, L, L] f32 (scores -> softmax in place)

  unsigned short* Qbf = (unsigned short*)d_ws;   // [B*L, H] bf16
  unsigned short* Kbf = Qbf + BLH;               // [B*L, H] bf16
  unsigned short* Vt  = Kbf + BLH;               // [B, H, L] bf16 (transposed)
  unsigned short* Pbf = Vt  + BLH;               // [B*L, L] bf16 probs

  const dim3 blk(256);
  const dim3 gProj(HID / BN, (BATCH * SEQ) / BM, 1);

  // Projections: x @ W^T + b, f32 in, bf16 out
  gemm_abT_kernel<float, float, 1><<<gProj, blk, 0, stream>>>(
      queries, Wq, bq, 1.0f, Qbf, HID, HID, HID, HID, 0, 0, 0);
  gemm_abT_kernel<float, float, 1><<<gProj, blk, 0, stream>>>(
      keys,    Wk, bk, 1.0f, Kbf, HID, HID, HID, HID, 0, 0, 0);
  gemm_abT_kernel<float, float, 2><<<gProj, blk, 0, stream>>>(
      values,  Wv, bv, 1.0f, Vt,  HID, HID, HID, 0,   0, 0, 0);

  // Scores: Q @ K^T * (1/sqrt(H)), per batch, f32 straight into d_out attn region
  gemm_abT_kernel<unsigned short, unsigned short, 0><<<dim3(SEQ / BN, SEQ / BM, BATCH), blk, 0, stream>>>(
      Qbf, Kbf, nullptr, 0.03125f, attnF, HID, HID, HID, SEQ,
      (long)SEQ * HID, (long)SEQ * HID, (long)SEQ * SEQ);

  // Softmax rows (in place) + bf16 probs for the context GEMM
  softmax_kernel<<<dim3(BATCH * SEQ), blk, 0, stream>>>(attnF, Pbf);

  // Context: P @ V  (via V^T, per batch), f32 out
  gemm_abT_kernel<unsigned short, unsigned short, 0><<<dim3(HID / BN, SEQ / BM, BATCH), blk, 0, stream>>>(
      Pbf, Vt, nullptr, 1.0f, ctx, SEQ, SEQ, SEQ, HID,
      (long)SEQ * SEQ, (long)HID * SEQ, (long)SEQ * HID);
}
